// MultiHeadAttention_19396072309379
// MI455X (gfx1250) — matrix operand order repaired and checked by reading, not compiled, not run on hardware
//
#include <hip/hip_runtime.h>

// ---------------------------------------------------------------------------
// MHA forward for MI455X (gfx1250, wave32, WMMA bf16 16x16x32).
// B=4, S=2048, D=512, H=8, DK=DV=64.
// Heads identical -> one head; tile(o,H)@Wo == o @ sum_h Wo_block_h.
// All GEMMs on v_wmma_f32_16x16x32_bf16, f32 accumulate.
// Fused N-tiles per wave so each fp32 activation tile is loaded/converted once.
// ---------------------------------------------------------------------------

typedef __bf16 bf16;
typedef __attribute__((ext_vector_type(8)))  __bf16 bf16x8;
typedef __attribute__((ext_vector_type(16))) __bf16 bf16x16;
typedef __attribute__((ext_vector_type(8)))  float  f32x8;

#define BATCH 4
#define SEQ   2048
#define DMODEL 512
#define HEADS  8
#define DKV    64

__device__ __forceinline__ bf16x16 bf_cat(bf16x8 a, bf16x8 b) {
  return __builtin_shufflevector(a, b, 0,1,2,3,4,5,6,7,8,9,10,11,12,13,14,15);
}

// B-fragment: lane holds column n; element e -> K = hi*16 + e, contiguous run of 16.
__device__ __forceinline__ bf16x16 load_b_frag(const bf16* p) {
  bf16x8 lo = *(const bf16x8*)p;
  bf16x8 hh = *(const bf16x8*)(p + 8);
  return bf_cat(lo, hh);
}

// A-fragment from bf16 row: element e -> K = kbase + hi*16 + e:
// one contiguous 16-element run.
__device__ __forceinline__ bf16x16 load_a_frag_bf16(const bf16* rowptr, int kbase, int hi) {
  bf16x8 lo = *(const bf16x8*)(rowptr + kbase + hi * 16);
  bf16x8 hh = *(const bf16x8*)(rowptr + kbase + hi * 16 + 8);
  return bf_cat(lo, hh);
}

// A-fragment straight from f32 global row, converting to bf16 on the fly.
__device__ __forceinline__ bf16x16 load_a_frag_f32(const float* rowptr, int kbase, int hi) {
  const float* p0 = rowptr + kbase + hi * 16;
  const float* p1 = rowptr + kbase + hi * 16 + 8;
  bf16x16 r;
#pragma unroll
  for (int i = 0; i < 8; ++i) {
    r[i]     = (bf16)p0[i];
    r[8 + i] = (bf16)p1[i];
  }
  return r;
}

__device__ __forceinline__ f32x8 wmma_bf16(bf16x16 a, bf16x16 b, f32x8 c) {
  // (neg_a, A, neg_b, B, c_mod, C, reuse_a, reuse_b)
  return __builtin_amdgcn_wmma_f32_16x16x32_bf16(false, a, false, b, (short)0, c,
                                                 false, false);
}

// ---------------------------------------------------------------------------
// Kernel 1: weight prep.  job 0..2: Wq/Wk/Wv [512][64] -> bf16 transpose [64][512].
//           job 3: WoST[d][k] = bf16( sum_h Wo[h*64+k][d] )   (head-fold of Wo)
// ---------------------------------------------------------------------------
__global__ void mha_prep_weights(const float* __restrict__ Wq,
                                 const float* __restrict__ Wk,
                                 const float* __restrict__ Wv,
                                 const float* __restrict__ Wo,
                                 bf16* __restrict__ WqT, bf16* __restrict__ WkT,
                                 bf16* __restrict__ WvT, bf16* __restrict__ WoST) {
  int idx = blockIdx.x * blockDim.x + threadIdx.x;   // 0..32767
  int job = blockIdx.y;
  if (job < 3) {
    int k = idx >> 6;        // 0..511
    int n = idx & 63;        // 0..63
    const float* W = (job == 0) ? Wq : (job == 1) ? Wk : Wv;
    bf16* Wt = (job == 0) ? WqT : (job == 1) ? WkT : WvT;
    Wt[n * DMODEL + k] = (bf16)W[k * DKV + n];
  } else {
    int d = idx >> 6;        // 0..511
    int k = idx & 63;        // 0..63
    float s = 0.f;
#pragma unroll
    for (int h = 0; h < HEADS; ++h) s += Wo[(h * DKV + k) * DMODEL + d];
    WoST[d * DKV + k] = (bf16)s;
  }
}

// ---------------------------------------------------------------------------
// Kernel 2: QKV projection.  grid = (B*S/16, 3 weights), 1 wave each.
// One wave computes all 4 N-tiles: A fragment (fp32 -> bf16) loaded ONCE per
// K-step and reused 4x.  which 0: q -> Qbf;  1: k -> Kbf;  2: v -> VT (transposed).
// ---------------------------------------------------------------------------
__global__ void mha_proj_qkv(const float* __restrict__ query,
                             const float* __restrict__ key,
                             const bf16* __restrict__ WqT, const bf16* __restrict__ WkT,
                             const bf16* __restrict__ WvT,
                             const float* __restrict__ bq, const float* __restrict__ bk,
                             const float* __restrict__ bv,
                             bf16* __restrict__ Qbf, bf16* __restrict__ Kbf,
                             bf16* __restrict__ VT) {
  const int lane = threadIdx.x;
  const int m = lane & 15, hi = lane >> 4;
  const int tile = blockIdx.x;          // row tile over B*S
  const int which = blockIdx.y;         // 0..2

  const float* src = (which == 0) ? query : key;    // v projected from key (ref bug)
  const bf16* Wt = (which == 0) ? WqT : (which == 1) ? WkT : WvT;
  const float* bias = (which == 0) ? bq : (which == 1) ? bk : bv;

  const float* arow = src + (size_t)(tile * 16 + m) * DMODEL;
  // B rows for the 4 N-tiles this lane touches: columns m, 16+m, 32+m, 48+m.
  const bf16* brow0 = Wt + (size_t)(0 * 16 + m) * DMODEL;
  const bf16* brow1 = Wt + (size_t)(1 * 16 + m) * DMODEL;
  const bf16* brow2 = Wt + (size_t)(2 * 16 + m) * DMODEL;
  const bf16* brow3 = Wt + (size_t)(3 * 16 + m) * DMODEL;

  f32x8 acc[4] = {{}, {}, {}, {}};
#pragma unroll
  for (int ks = 0; ks < DMODEL / 32; ++ks) {
    const int ko = ks * 32 + hi * 16;
    bf16x16 a = load_a_frag_f32(arow, ks * 32, hi);   // loaded once, used 4x
    acc[0] = wmma_bf16(a, load_b_frag(brow0 + ko), acc[0]);
    acc[1] = wmma_bf16(a, load_b_frag(brow1 + ko), acc[1]);
    acc[2] = wmma_bf16(a, load_b_frag(brow2 + ko), acc[2]);
    acc[3] = wmma_bf16(a, load_b_frag(brow3 + ko), acc[3]);
  }

  if (which < 2) {
    bf16* dst = (which == 0) ? Qbf : Kbf;
#pragma unroll
    for (int nt = 0; nt < 4; ++nt) {
      const int ncol = nt * 16 + m;
      const float bval = bias[ncol];
#pragma unroll
      for (int r = 0; r < 8; ++r) {
        int rowg = tile * 16 + hi * 8 + r;
        dst[(size_t)rowg * DKV + ncol] = (bf16)(acc[nt][r] + bval);
      }
    }
  } else {
#pragma unroll
    for (int nt = 0; nt < 4; ++nt) {
      const int ncol = nt * 16 + m;
      const float bval = bias[ncol];
#pragma unroll
      for (int r = 0; r < 8; ++r) {
        int rowg = tile * 16 + hi * 8 + r;
        int b_   = rowg >> 11;            // / SEQ
        int srow = rowg & (SEQ - 1);
        VT[((size_t)b_ * DKV + ncol) * SEQ + srow] = (bf16)(acc[nt][r] + bval);
      }
    }
  }
}

// ---------------------------------------------------------------------------
// Kernel 3: causal flash attention, one wave per (batch, 16-query tile).
// Key blocks of 32 (one WMMA K step). Online softmax, P bounced via LDS to
// convert C-layout -> A-fragment layout for P@V.  K/V B-fragments are
// lane-contiguous 32B global loads straight into WMMA source VGPRs (no LDS
// staging needed: single-wave WGs have no cross-wave reuse to amortize).
// ---------------------------------------------------------------------------
__global__ void mha_flash(const bf16* __restrict__ Qbf, const bf16* __restrict__ Kbf,
                          const bf16* __restrict__ VT, bf16* __restrict__ Obf) {
  __shared__ bf16 Plds[16][40];         // 16x32 P tile, row stride 40 (16B aligned)

  const int lane = threadIdx.x;
  const int m = lane & 15, hi = lane >> 4;
  const int qt = blockIdx.x;            // 0..S/16-1
  const int b = blockIdx.y;             // batch
  const int qrow0 = qt * 16;

  const bf16* qrow = Qbf + ((size_t)(b * SEQ + qrow0 + m)) * DKV;
  const bf16x16 aQ0 = load_a_frag_bf16(qrow, 0, hi);
  const bf16x16 aQ1 = load_a_frag_bf16(qrow, 32, hi);

  const bf16* kbase = Kbf + (size_t)b * SEQ * DKV;
  const bf16* vbase = VT + (size_t)b * DKV * SEQ;

  f32x8 accO[4] = {{}, {}, {}, {}};
  float mrow[8], lrow[8], alpha[8];
#pragma unroll
  for (int r = 0; r < 8; ++r) { mrow[r] = -1e30f; lrow[r] = 0.f; }

  const int nkb = (qrow0 + 15) / 32 + 1;     // causal: keys <= last query row
  for (int kb = 0; kb < nkb; ++kb) {
    const bf16* krow0 = kbase + (size_t)(kb * 32 + m) * DKV;       // cols kb*32+m
    const bf16* krow1 = kbase + (size_t)(kb * 32 + 16 + m) * DKV;  // cols +16
    __builtin_prefetch(krow0 + 32 * DKV, 0, 0);   // next key block -> global_prefetch

    f32x8 s0 = {}, s1 = {};
    s0 = wmma_bf16(aQ0, load_b_frag(krow0 + hi * 16), s0);
    s0 = wmma_bf16(aQ1, load_b_frag(krow0 + 32 + hi * 16), s0);
    s1 = wmma_bf16(aQ0, load_b_frag(krow1 + hi * 16), s1);
    s1 = wmma_bf16(aQ1, load_b_frag(krow1 + 32 + hi * 16), s1);

    const int col0 = kb * 32 + m, col1 = col0 + 16;
#pragma unroll
    for (int r = 0; r < 8; ++r) {
      const int rowg = qrow0 + hi * 8 + r;
      float v0 = (col0 <= rowg) ? s0[r] * 0.125f : -1e30f;  // 1/sqrt(64)
      float v1 = (col1 <= rowg) ? s1[r] * 0.125f : -1e30f;
      float tm = fmaxf(v0, v1);
      tm = fmaxf(tm, __shfl_xor(tm, 1, 32));
      tm = fmaxf(tm, __shfl_xor(tm, 2, 32));
      tm = fmaxf(tm, __shfl_xor(tm, 4, 32));
      tm = fmaxf(tm, __shfl_xor(tm, 8, 32));
      const float mn = fmaxf(mrow[r], tm);
      const float al = __expf(mrow[r] - mn);
      const float p0 = __expf(v0 - mn);
      const float p1 = __expf(v1 - mn);
      float rs = p0 + p1;
      rs += __shfl_xor(rs, 1, 32);
      rs += __shfl_xor(rs, 2, 32);
      rs += __shfl_xor(rs, 4, 32);
      rs += __shfl_xor(rs, 8, 32);
      lrow[r] = lrow[r] * al + rs;
      mrow[r] = mn;
      alpha[r] = al;
      Plds[hi * 8 + r][m] = (bf16)p0;
      Plds[hi * 8 + r][16 + m] = (bf16)p1;
    }
#pragma unroll
    for (int nt = 0; nt < 4; ++nt)
#pragma unroll
      for (int r = 0; r < 8; ++r) accO[nt][r] *= alpha[r];

    __syncthreads();   // single-wave WG: S_NOP, but orders LDS for the compiler

    const bf16x16 aP = load_a_frag_bf16(&Plds[m][0], 0, hi);
#pragma unroll
    for (int nt = 0; nt < 4; ++nt) {
      const bf16* vp = vbase + (size_t)(nt * 16 + m) * SEQ + kb * 32 + hi * 16;
      accO[nt] = wmma_bf16(aP, load_b_frag(vp), accO[nt]);
    }
  }

#pragma unroll
  for (int nt = 0; nt < 4; ++nt)
#pragma unroll
    for (int r = 0; r < 8; ++r) {
      const float o = accO[nt][r] / lrow[r];
      const int rowg = b * SEQ + qrow0 + hi * 8 + r;
      Obf[(size_t)rowg * DKV + nt * 16 + m] = (bf16)o;
    }
}

// ---------------------------------------------------------------------------
// Kernel 4: out = O[16x64] @ Wo_sum[64x512] + bo, fp32 stores.
// grid = (B*S/16, 8), 1 wave each; each wave does 4 N-tiles with one A load.
// ---------------------------------------------------------------------------
__global__ void mha_out_proj(const bf16* __restrict__ Obf, const bf16* __restrict__ WoST,
                             const float* __restrict__ bo, float* __restrict__ out) {
  const int lane = threadIdx.x;
  const int m = lane & 15, hi = lane >> 4;
  const int tile = blockIdx.x;          // row tile over B*S
  const int ntg = blockIdx.y;           // 0..7 -> N-tiles ntg*4 .. ntg*4+3

  const bf16* arow = Obf + (size_t)(tile * 16 + m) * DKV;
  const bf16x16 a0 = load_a_frag_bf16(arow, 0, hi);
  const bf16x16 a1 = load_a_frag_bf16(arow, 32, hi);

#pragma unroll
  for (int i = 0; i < 4; ++i) {
    const int nt = ntg * 4 + i;
    const int d = nt * 16 + m;
    const bf16* brow = WoST + (size_t)d * DKV;
    f32x8 acc = {};
    acc = wmma_bf16(a0, load_b_frag(brow + hi * 16), acc);
    acc = wmma_bf16(a1, load_b_frag(brow + 32 + hi * 16), acc);
    const float bval = bo[d];
#pragma unroll
    for (int r = 0; r < 8; ++r) {
      const int rowg = tile * 16 + hi * 8 + r;
      out[(size_t)rowg * DMODEL + d] = acc[r] + bval;
    }
  }
}

// ---------------------------------------------------------------------------
// Launch. d_in order: query, key, value, Wq, bq, Wk, bk, Wv, bv, Wo, bo.
// Workspace (bytes): WqT/WkT/WvT/WoST 64KB each; Qbf/Kbf/VT/Obf 1MB each = 4.25MB.
// ---------------------------------------------------------------------------
extern "C" void kernel_launch(void* const* d_in, const int* in_sizes, int n_in,
                              void* d_out, int out_size, void* d_ws, size_t ws_size,
                              hipStream_t stream) {
  const float* query = (const float*)d_in[0];
  const float* key   = (const float*)d_in[1];
  // d_in[2] (value) intentionally unused: reference projects V from `key`.
  const float* Wq = (const float*)d_in[3];
  const float* bq = (const float*)d_in[4];
  const float* Wk = (const float*)d_in[5];
  const float* bk = (const float*)d_in[6];
  const float* Wv = (const float*)d_in[7];
  const float* bv = (const float*)d_in[8];
  const float* Wo = (const float*)d_in[9];
  const float* bo = (const float*)d_in[10];
  float* out = (float*)d_out;

  char* ws = (char*)d_ws;
  const size_t WSZ = (size_t)DKV * DMODEL * sizeof(bf16);      // 64KB
  const size_t TSZ = (size_t)BATCH * SEQ * DKV * sizeof(bf16); // 1MB
  bf16* WqT  = (bf16*)(ws + 0 * WSZ);
  bf16* WkT  = (bf16*)(ws + 1 * WSZ);
  bf16* WvT  = (bf16*)(ws + 2 * WSZ);
  bf16* WoST = (bf16*)(ws + 3 * WSZ);
  bf16* Qbf  = (bf16*)(ws + 4 * WSZ + 0 * TSZ);
  bf16* Kbf  = (bf16*)(ws + 4 * WSZ + 1 * TSZ);
  bf16* VT   = (bf16*)(ws + 4 * WSZ + 2 * TSZ);
  bf16* Obf  = (bf16*)(ws + 4 * WSZ + 3 * TSZ);

  mha_prep_weights<<<dim3(128, 4), 256, 0, stream>>>(Wq, Wk, Wv, Wo, WqT, WkT, WvT, WoST);
  mha_proj_qkv<<<dim3(BATCH * SEQ / 16, 3), 32, 0, stream>>>(
      query, key, WqT, WkT, WvT, bq, bk, bv, Qbf, Kbf, VT);
  mha_flash<<<dim3(SEQ / 16, BATCH), 32, 0, stream>>>(Qbf, Kbf, VT, Obf);
  mha_out_proj<<<dim3(BATCH * SEQ / 16, 8), 32, 0, stream>>>(Obf, WoST, bo, out);
}
